// Net_68143951118849
// MI455X (gfx1250) — matrix operand order repaired and checked by reading, not compiled, not run on hardware
//
#include <hip/hip_runtime.h>
#include <hip/hip_bf16.h>

// ---------------------------------------------------------------------------
// GCN (3x GraphConv, norm='both') for MI455X / gfx1250 (wave32, WMMA).
// GEMMs run on v_wmma_f32_16x16x32_bf16; SpMM is edge-parallel float4
// gathers + global float atomics (L2-resident: feature matrices are 25.6MB
// vs 192MB L2). W is staged transposed in LDS so B fragments load as two
// aligned ds_load_b128 per fragment.
// ---------------------------------------------------------------------------

typedef __attribute__((ext_vector_type(16))) __bf16 v16bf;
typedef __attribute__((ext_vector_type(8)))  __bf16 v8bf;
typedef __attribute__((ext_vector_type(8)))  float  v8f;

// ---------------------------------------------------------------------------
// Utility kernels
// ---------------------------------------------------------------------------
__global__ void zero_kernel(float* __restrict__ p, int n) {
  int i = blockIdx.x * blockDim.x + threadIdx.x;
  if (i < n) p[i] = 0.0f;
}

__global__ void degree_kernel(const int* __restrict__ src, const int* __restrict__ dst,
                              float* __restrict__ outdeg, float* __restrict__ indeg, int E) {
  int e = blockIdx.x * blockDim.x + threadIdx.x;
  if (e < E) {
    atomicAdd(&outdeg[src[e]], 1.0f);
    atomicAdd(&indeg[dst[e]], 1.0f);
  }
}

__global__ void rsqrt_kernel(float* __restrict__ p, int n) {
  int i = blockIdx.x * blockDim.x + threadIdx.x;
  if (i < n) {
    float d = p[i];
    p[i] = rsqrtf(d < 1.0f ? 1.0f : d);
  }
}

// ---------------------------------------------------------------------------
// WMMA GEMM: out[r, c] = sum_k (in[r,k] * norm_out[r]) * W[k, c]
//   in:  [nrows, K] fp32, W: [K, NOUT_REAL] fp32 (zero-padded to NOUT_PAD)
//   out: [nrows, NOUT_PAD] fp32 (padded cols written as 0)
// One wave per 16-row tile. W is held transposed in LDS: Wl[c*KP + k],
// KP = K+8 bf16 so each lane's 16 contiguous K-values come back as two
// aligned ds_load_b128 and lanes spread across LDS banks (272B row pitch).
// ---------------------------------------------------------------------------
template <int K, int NOUT_PAD, int NOUT_REAL>
__global__ void gemm_wmma(const float* __restrict__ in, const float* __restrict__ W,
                          const float* __restrict__ norm_out, float* __restrict__ out,
                          int nrows) {
  constexpr int KP = K + 8;  // padded bf16 pitch: 272B (K=128) / 144B (K=64)
  __shared__ __bf16 Wl[NOUT_PAD * KP];

  // Cooperative transpose-load of W into LDS, 8 bf16 per LDS store.
  constexpr int KCH = K / 8;
  for (int idx = threadIdx.x; idx < NOUT_PAD * KCH; idx += blockDim.x) {
    const int c = idx / KCH;
    const int k0 = (idx % KCH) * 8;
    v8bf v;
#pragma unroll
    for (int j = 0; j < 8; ++j)
      v[j] = (c < NOUT_REAL) ? (__bf16)W[(k0 + j) * NOUT_REAL + c] : (__bf16)0.0f;
    *(v8bf*)&Wl[c * KP + k0] = v;
  }
  __syncthreads();

  const int wave = threadIdx.x >> 5;
  const int lane = threadIdx.x & 31;
  const int tile = blockIdx.x * (blockDim.x >> 5) + wave;
  const int ntiles = nrows >> 4;
  if (tile >= ntiles) return;  // wave-uniform exit: active waves keep EXEC full

  const int rbase = tile << 4;
  const int mrow = lane & 15;
  const int hi = (lane < 16) ? 0 : 1;
  const int r = rbase + mrow;
  const float s = norm_out[r];
  const float* __restrict__ row = in + (size_t)r * K;

  // A fragments: 16x32 bf16; lanes 0-15 hold K 0..15, lanes 16-31
  // hold K 16..31 (matches B fragment K order).
  constexpr int KT = K / 32;
  v16bf afrag[KT];
#pragma unroll
  for (int kt = 0; kt < KT; ++kt) {
    const int kb = kt * 32 + hi * 16;
#pragma unroll
    for (int j = 0; j < 8; ++j) afrag[kt][j] = (__bf16)(row[kb + j] * s);
#pragma unroll
    for (int j = 0; j < 8; ++j) afrag[kt][8 + j] = (__bf16)(row[kb + 8 + j] * s);
  }

#pragma unroll
  for (int ct = 0; ct < NOUT_PAD / 16; ++ct) {
    v8f acc = {0.f, 0.f, 0.f, 0.f, 0.f, 0.f, 0.f, 0.f};
    const int c = ct * 16 + mrow;  // output column for this lane
#pragma unroll
    for (int kt = 0; kt < KT; ++kt) {
      // B fragment: 32x16 bf16; lanes 0-15 hold K 0..15 of column `c`,
      // lanes 16-31 hold K 16..31. Contiguous in transposed LDS image:
      // two aligned 16B chunks.
      const int kb = kt * 32 + hi * 16;
      const __bf16* bptr = &Wl[c * KP + kb];
      v8bf blo = *(const v8bf*)(bptr);
      v8bf bhi = *(const v8bf*)(bptr + 8);
      v16bf bfrag = __builtin_shufflevector(blo, bhi, 0, 1, 2, 3, 4, 5, 6, 7,
                                            8, 9, 10, 11, 12, 13, 14, 15);
      acc = __builtin_amdgcn_wmma_f32_16x16x32_bf16(
          /*neg_a=*/false, afrag[kt], /*neg_b=*/false, bfrag,
          /*c_mod=*/(short)0, acc, /*reuse_a=*/false, /*reuse_b=*/false);
    }
    // D layout: VGPR v -> row (v + 8*hi), col = lane&15 within the tile.
    const int rowo = rbase + hi * 8;
#pragma unroll
    for (int v = 0; v < 8; ++v)
      out[(size_t)(rowo + v) * NOUT_PAD + c] = acc[v];
  }
}

// Scalar tail fallback for nrows % 16 != 0 (not hit for N=100000).
template <int K, int NOUT_PAD, int NOUT_REAL>
__global__ void gemm_tail(const float* __restrict__ in, const float* __restrict__ W,
                          const float* __restrict__ norm_out, float* __restrict__ out,
                          int rstart, int nrows) {
  int i = blockIdx.x * blockDim.x + threadIdx.x;
  int r = rstart + i / NOUT_PAD;
  int c = i % NOUT_PAD;
  if (r >= nrows) return;
  float acc = 0.0f;
  if (c < NOUT_REAL) {
    float s = norm_out[r];
    for (int k = 0; k < K; ++k)
      acc += (in[(size_t)r * K + k] * s) * W[k * NOUT_REAL + c];
  }
  out[(size_t)r * NOUT_PAD + c] = acc;
}

// ---------------------------------------------------------------------------
// SpMM: agg[dst] += m[src] per edge. One thread per (edge, 4-feature chunk).
// float4 gather + 4 packed global_atomic_add_f32 (stays in L2 on MI455X).
// ---------------------------------------------------------------------------
template <int LD>
__global__ void spmm_kernel(const float* __restrict__ m, const int* __restrict__ src,
                            const int* __restrict__ dst, float* __restrict__ agg, int E) {
  constexpr int CH = LD / 4;
  int tid = blockIdx.x * blockDim.x + threadIdx.x;
  int e = tid / CH;
  int j = tid % CH;
  if (e >= E) return;
  int s = src[e];
  int d = dst[e];
  const float4 v = ((const float4*)(m + (size_t)s * LD))[j];
  float* o = agg + (size_t)d * LD + j * 4;
  atomicAdd(o + 0, v.x);
  atomicAdd(o + 1, v.y);
  atomicAdd(o + 2, v.z);
  atomicAdd(o + 3, v.w);
}

// ---------------------------------------------------------------------------
// Epilogue: out[n, f] = (relu?)(agg[n, f] * norm_in[n] + bias[f])
// ---------------------------------------------------------------------------
__global__ void post_kernel(const float* __restrict__ agg, const float* __restrict__ norm_in,
                            const float* __restrict__ bias, float* __restrict__ out,
                            int n, int ld_in, int ld_out, int F, int do_relu) {
  int i = blockIdx.x * blockDim.x + threadIdx.x;
  if (i >= n * F) return;
  int node = i / F;
  int f = i % F;
  float v = agg[(size_t)node * ld_in + f] * norm_in[node] + bias[f];
  if (do_relu) v = v > 0.0f ? v : 0.0f;
  out[(size_t)node * ld_out + f] = v;
}

// ---------------------------------------------------------------------------
// Launch
// ---------------------------------------------------------------------------
extern "C" void kernel_launch(void* const* d_in, const int* in_sizes, int n_in,
                              void* d_out, int out_size, void* d_ws, size_t ws_size,
                              hipStream_t stream) {
  const float* x   = (const float*)d_in[0];
  const float* W0  = (const float*)d_in[1];
  const float* b0  = (const float*)d_in[2];
  const float* W1  = (const float*)d_in[3];
  const float* b1  = (const float*)d_in[4];
  const float* W2  = (const float*)d_in[5];
  const float* b2  = (const float*)d_in[6];
  const int* esrc  = (const int*)d_in[7];
  const int* edst  = (const int*)d_in[8];

  const int Nn = in_sizes[0] / 128;  // nodes
  const int E  = in_sizes[7];        // edges

  float* ws       = (float*)d_ws;
  float* norm_out = ws;                       // [Nn] (deg accumulator then norm)
  float* norm_in  = ws + Nn;                  // [Nn]
  float* m        = ws + (size_t)2 * Nn;      // [Nn, 64] (layer2 uses ld=48)
  float* agg      = m + (size_t)64 * Nn;      // [Nn, 64]
  float* h        = agg + (size_t)64 * Nn;    // [Nn, 64]

  auto cdiv = [](long a, long b) { return (int)((a + b - 1) / b); };

  // ---- degree norms ----
  zero_kernel<<<cdiv(2L * Nn, 256), 256, 0, stream>>>(norm_out, 2 * Nn);
  degree_kernel<<<cdiv(E, 256), 256, 0, stream>>>(esrc, edst, norm_out, norm_in, E);
  rsqrt_kernel<<<cdiv(2L * Nn, 256), 256, 0, stream>>>(norm_out, 2 * Nn);

  const int ntiles = Nn >> 4;
  const int tail = Nn & 15;
  const int gblocks = cdiv(ntiles, 4);  // 4 waves / block, 1 row-tile / wave

  // ---- layer 0: IN=128 -> H=64, relu ----
  gemm_wmma<128, 64, 64><<<gblocks, 128, 0, stream>>>(x, W0, norm_out, m, Nn);
  if (tail)
    gemm_tail<128, 64, 64><<<cdiv((long)tail * 64, 256), 256, 0, stream>>>(
        x, W0, norm_out, m, ntiles << 4, Nn);
  zero_kernel<<<cdiv((long)Nn * 64, 256), 256, 0, stream>>>(agg, Nn * 64);
  spmm_kernel<64><<<cdiv((long)E * 16, 256), 256, 0, stream>>>(m, esrc, edst, agg, E);
  post_kernel<<<cdiv((long)Nn * 64, 256), 256, 0, stream>>>(agg, norm_in, b0, h,
                                                            Nn, 64, 64, 64, 1);

  // ---- layer 1: H=64 -> H=64, relu ----
  gemm_wmma<64, 64, 64><<<gblocks, 128, 0, stream>>>(h, W1, norm_out, m, Nn);
  if (tail)
    gemm_tail<64, 64, 64><<<cdiv((long)tail * 64, 256), 256, 0, stream>>>(
        h, W1, norm_out, m, ntiles << 4, Nn);
  zero_kernel<<<cdiv((long)Nn * 64, 256), 256, 0, stream>>>(agg, Nn * 64);
  spmm_kernel<64><<<cdiv((long)E * 16, 256), 256, 0, stream>>>(m, esrc, edst, agg, E);
  post_kernel<<<cdiv((long)Nn * 64, 256), 256, 0, stream>>>(agg, norm_in, b1, h,
                                                            Nn, 64, 64, 64, 1);

  // ---- layer 2: H=64 -> C=40 (padded to 48), no relu ----
  gemm_wmma<64, 48, 40><<<gblocks, 128, 0, stream>>>(h, W2, norm_out, m, Nn);
  if (tail)
    gemm_tail<64, 48, 40><<<cdiv((long)tail * 48, 256), 256, 0, stream>>>(
        h, W2, norm_out, m, ntiles << 4, Nn);
  zero_kernel<<<cdiv((long)Nn * 48, 256), 256, 0, stream>>>(agg, Nn * 48);
  spmm_kernel<48><<<cdiv((long)E * 12, 256), 256, 0, stream>>>(m, esrc, edst, agg, E);
  post_kernel<<<cdiv((long)Nn * 40, 256), 256, 0, stream>>>(agg, norm_in, b2,
                                                            (float*)d_out, Nn, 48, 40, 40, 0);
}
